// SDCN_GNN_704374636672
// MI455X (gfx1250) — matrix operand order repaired and checked by reading, not compiled, not run on hardware
//
#include <hip/hip_runtime.h>
#include <hip/hip_bf16.h>
#include <stdint.h>

// ---------------------------------------------------------------------------
// Types for CDNA5 WMMA (wave32, 16x16x32 bf16 -> f32)
// ---------------------------------------------------------------------------
typedef __attribute__((ext_vector_type(16))) __bf16 v16bf;
typedef __attribute__((ext_vector_type(8)))  float  v8f;

union FragBF16 {
    uint4 q[2];   // 32 bytes = 16 bf16
    v16bf v;
};

__device__ __forceinline__ unsigned short f32_to_bf16(float f) {
    unsigned int u = __float_as_uint(f);
    u += 0x7FFFu + ((u >> 16) & 1u);   // round-to-nearest-even
    return (unsigned short)(u >> 16);
}

// ---------------------------------------------------------------------------
// Fused (ReLU -> (1-s)*h + s*tra) -> bf16, with zero K-padding.
// do_mix==0: plain f32 -> bf16 convert of h (no ReLU).
// ---------------------------------------------------------------------------
__global__ void mix_convert_bf16(const float* __restrict__ h,
                                 const float* __restrict__ tra,
                                 unsigned short* __restrict__ outA,
                                 int Nrows, int K, int Kp, int Hstride,
                                 int do_mix, float sigma) {
    size_t idx = (size_t)blockIdx.x * blockDim.x + threadIdx.x;
    size_t total = (size_t)Nrows * (size_t)Kp;
    if (idx >= total) return;
    int    k = (int)(idx % (size_t)Kp);
    size_t r = idx / (size_t)Kp;
    float  v = 0.0f;
    if (k < K) {
        if (do_mix) {
            float hv = h[r * (size_t)Hstride + k];
            hv = hv > 0.0f ? hv : 0.0f;                       // ReLU on previous layer
            v = (1.0f - sigma) * hv + sigma * tra[r * (size_t)K + k];
        } else {
            v = h[r * (size_t)Hstride + k];
        }
    }
    outA[idx] = f32_to_bf16(v);
}

// ---------------------------------------------------------------------------
// Fused (ReLU -> (1-s)*h + s*tra) -> f32 (for the spmm-first layer),
// zero K-padding so the padded columns are clean.
// ---------------------------------------------------------------------------
__global__ void mix_f32(const float* __restrict__ h,
                        const float* __restrict__ tra,
                        float* __restrict__ outF,
                        int Nrows, int K, int Kp, int Hstride, float sigma) {
    size_t idx = (size_t)blockIdx.x * blockDim.x + threadIdx.x;
    size_t total = (size_t)Nrows * (size_t)Kp;
    if (idx >= total) return;
    int    k = (int)(idx % (size_t)Kp);
    size_t r = idx / (size_t)Kp;
    float  v = 0.0f;
    if (k < K) {
        float hv = h[r * (size_t)Hstride + k];
        hv = hv > 0.0f ? hv : 0.0f;
        v = (1.0f - sigma) * hv + sigma * tra[r * (size_t)K + k];
    }
    outF[idx] = v;
}

// ---------------------------------------------------------------------------
// Repack W (K x M, row-major f32) into bf16 blocked layout:
//   wp[(kb*Mp + col)*32 + (k & 31)]  for kb = k >> 5
// so each lane's B fragment (K 8h..8h+7 and 16+8h..16+8h+7 of one column) is two runs of 16 contiguous
// bytes. Pads (k>=K || col>=M) are zero.
// ---------------------------------------------------------------------------
__global__ void pack_weight_bf16(const float* __restrict__ W,
                                 unsigned short* __restrict__ wp,
                                 int K, int M, int Kp, int Mp) {
    int idx = blockIdx.x * blockDim.x + threadIdx.x;
    if (idx >= Kp * Mp) return;
    int k = idx / Mp, c = idx % Mp;
    float v = (k < K && c < M) ? W[(size_t)k * M + c] : 0.0f;
    wp[((size_t)(k >> 5) * Mp + c) * 32 + (k & 31)] = f32_to_bf16(v);
}

// ---------------------------------------------------------------------------
// GEMM: Out(N x Mp, f32) = A(N x Kp, bf16) * Wp(packed bf16)
// Block = 256 threads = 8 waves; wave computes a 16x64 tile (one A fragment
// shared by 4 WMMA accumulators). Block covers 128 rows x 64 cols.
// Fragment layouts per CDNA5 ISA 7.12.2 (see round-0 derivation).
// ---------------------------------------------------------------------------
__global__ void __launch_bounds__(256)
gemm_bf16_wmma(const unsigned short* __restrict__ A,
               const unsigned short* __restrict__ Wp,
               float* __restrict__ Out,
               int Nrows, int Kp, int Mp) {
    const int lane = threadIdx.x & 31;
    const int wave = threadIdx.x >> 5;
    const int hi   = lane >> 4;
    const int m15  = lane & 15;

    const int row0 = blockIdx.y * 128 + wave * 16;
    if (row0 >= Nrows) return;                       // wave-uniform: EXEC stays full
    const int col0 = blockIdx.x * 64;

    const unsigned short* arow = A + (size_t)(row0 + m15) * Kp + hi * 8;
    const unsigned short* wcol = Wp + ((size_t)(col0 + m15)) * 32 + hi * 8;
    const size_t wkb = (size_t)Mp * 32;              // packed stride per 32-K block

    v8f acc0 = {}, acc1 = {}, acc2 = {}, acc3 = {};

    for (int k0 = 0; k0 < Kp; k0 += 32) {
        FragBF16 a;
        a.q[0] = *(const uint4*)(arow + k0);         // K = k0+hi*8   .. +7
        a.q[1] = *(const uint4*)(arow + k0 + 16);    // K = k0+16+hi*8.. +7

        const unsigned short* wb = wcol + (size_t)(k0 >> 5) * wkb;
        FragBF16 b0, b1, b2, b3;
        b0.q[0] = *(const uint4*)(wb +    0); b0.q[1] = *(const uint4*)(wb +   16);
        b1.q[0] = *(const uint4*)(wb +  512); b1.q[1] = *(const uint4*)(wb +  528);
        b2.q[0] = *(const uint4*)(wb + 1024); b2.q[1] = *(const uint4*)(wb + 1040);
        b3.q[0] = *(const uint4*)(wb + 1536); b3.q[1] = *(const uint4*)(wb + 1552);

        acc0 = __builtin_amdgcn_wmma_f32_16x16x32_bf16(false, a.v, false, b0.v, (short)0, acc0, false, false);
        acc1 = __builtin_amdgcn_wmma_f32_16x16x32_bf16(false, a.v, false, b1.v, (short)0, acc1, false, false);
        acc2 = __builtin_amdgcn_wmma_f32_16x16x32_bf16(false, a.v, false, b2.v, (short)0, acc2, false, false);
        acc3 = __builtin_amdgcn_wmma_f32_16x16x32_bf16(false, a.v, false, b3.v, (short)0, acc3, false, false);
    }

    float* ob = Out + (size_t)(row0 + hi * 8) * Mp + col0 + m15;
#pragma unroll
    for (int r = 0; r < 8; ++r) {
        ob[(size_t)r * Mp +  0] = acc0[r];
        ob[(size_t)r * Mp + 16] = acc1[r];
        ob[(size_t)r * Mp + 32] = acc2[r];
        ob[(size_t)r * Mp + 48] = acc3[r];
    }
}

// ---------------------------------------------------------------------------
// Zero fill (grid-stride)
// ---------------------------------------------------------------------------
__global__ void zero_f32(float* __restrict__ p, size_t n) {
    size_t i = (size_t)blockIdx.x * blockDim.x + threadIdx.x;
    size_t stride = (size_t)gridDim.x * blockDim.x;
    for (; i < n; i += stride) p[i] = 0.0f;
}

// ---------------------------------------------------------------------------
// SpMM scatter: out[row[e], j] += vals[e] * m[col[e], j]
// Block (32,8): each wave covers one edge x 32 consecutive features ->
// coalesced row gathers (L2-resident source) + coalesced f32 atomics.
// ---------------------------------------------------------------------------
__global__ void spmm_scatter(const float* __restrict__ vals,
                             const int* __restrict__ rows,
                             const int* __restrict__ cols,
                             const float* __restrict__ Mmat,
                             float* __restrict__ Out,
                             int E, int Mcols, int Mp) {
    int e = blockIdx.x * blockDim.y + threadIdx.y;
    int j = blockIdx.y * 32 + threadIdx.x;
    if (e >= E || j >= Mcols) return;
    float v = vals[e];
    int r = rows[e], c = cols[e];
    atomicAdd(&Out[(size_t)r * Mp + j], v * Mmat[(size_t)c * Mp + j]);
}

// ---------------------------------------------------------------------------
// Row softmax over 10 classes
// ---------------------------------------------------------------------------
__global__ void softmax10(const float* __restrict__ H, float* __restrict__ out,
                          int Nrows, int Mp) {
    int r = blockIdx.x * blockDim.x + threadIdx.x;
    if (r >= Nrows) return;
    float v[10];
    float mx = -3.4e38f;
#pragma unroll
    for (int j = 0; j < 10; ++j) { v[j] = H[(size_t)r * Mp + j]; mx = fmaxf(mx, v[j]); }
    float s = 0.0f;
#pragma unroll
    for (int j = 0; j < 10; ++j) { v[j] = __expf(v[j] - mx); s += v[j]; }
    float inv = 1.0f / s;
#pragma unroll
    for (int j = 0; j < 10; ++j) out[(size_t)r * 10 + j] = v[j] * inv;
}

// ---------------------------------------------------------------------------
// Host orchestration
// ---------------------------------------------------------------------------
extern "C" void kernel_launch(void* const* d_in, const int* in_sizes, int n_in,
                              void* d_out, int out_size, void* d_ws, size_t ws_size,
                              hipStream_t stream) {
    const float* x     = (const float*)d_in[0];
    const float* tra1  = (const float*)d_in[1];
    const float* tra2  = (const float*)d_in[2];
    const float* tra3  = (const float*)d_in[3];
    const float* zin   = (const float*)d_in[4];
    const float* evals = (const float*)d_in[5];
    const int*   erow  = (const int*)d_in[6];
    const int*   ecol  = (const int*)d_in[7];
    const float* W[5]  = {(const float*)d_in[8],  (const float*)d_in[9],
                          (const float*)d_in[10], (const float*)d_in[11],
                          (const float*)d_in[12]};

    const int Nn = 50000, E = 1600000;
    const int Kd[5] = {512, 500, 500, 2000, 10};     // GEMM K per layer
    const int Md[5] = {500, 500, 2000, 10, 10};      // GEMM M per layer
    const int Kp[5] = {512, 512, 512, 2016, 32};     // K padded to mult of 32
    const int Mp[5] = {512, 512, 2048, 64, 64};      // M padded to mult of 64
    const float* mixsrc[5] = {nullptr, tra1, tra2, tra3, zin};

    // Workspace layout (256B-aligned offsets):
    //   bufA  : bf16 activations, N*2016*2               = 201.6 MB
    //   bufM  : f32 GEMM output / spmm-first scratch      = 409.6 MB
    //   O0/O1 : f32 ping-pong layer outputs, 409.6 MB each
    //   wp    : packed bf16 weights (<= 2 MB)
    char* w = (char*)d_ws;
    unsigned short* bufA = (unsigned short*)(w);
    float*          bufM = (float*)(w + 208000000ull);
    float*          O0   = (float*)(w + 624000000ull);
    float*          O1   = (float*)(w + 1040000000ull);
    unsigned short* wp   = (unsigned short*)(w + 1456000000ull);
    float* Obuf[2] = {O0, O1};

    const float* hprev = x;
    int hstride = 512;

    for (int l = 0; l < 5; ++l) {
        float* Oc = Obuf[l & 1];

        // Weights repacked for WMMA either way.
        int wtot = Kp[l] * Mp[l];
        pack_weight_bf16<<<(wtot + 255) / 256, 256, 0, stream>>>(
            W[l], wp, Kd[l], Md[l], Kp[l], Mp[l]);

        if (l == 2) {
            // -------- spmm-first variant (K=500 < M=2000): --------
            //   O = spmm(A, mix) @ W3   ==   spmm(A, mix @ W3)
            // spmm runs at width 500 instead of 2000 (4x less atomic traffic)
            // and the gathered matrix (102 MB) stays resident in L2.
            float* bufF = bufM;                        // N x 512 f32 mixed input
            float* bufS = bufM + (size_t)Nn * 512;     // N x 512 f32 spmm result
            size_t ftot = (size_t)Nn * 512;

            mix_f32<<<(unsigned)((ftot + 255) / 256), 256, 0, stream>>>(
                hprev, mixsrc[l], bufF, Nn, Kd[l], 512, hstride, 0.5f);

            zero_f32<<<8192, 256, 0, stream>>>(bufS, ftot);
            dim3 sg((unsigned)((E + 7) / 8), (unsigned)((Kd[l] + 31) / 32));
            spmm_scatter<<<sg, dim3(32, 8), 0, stream>>>(
                evals, erow, ecol, bufF, bufS, E, Kd[l], 512);

            // Straight f32 -> bf16 convert of spmm result (pads already zero).
            mix_convert_bf16<<<(unsigned)((ftot + 255) / 256), 256, 0, stream>>>(
                bufS, nullptr, bufA, Nn, 512, 512, 512, 0, 0.5f);

            // GEMM writes the layer output directly (dense, no atomics).
            dim3 gg((unsigned)(Mp[l] / 64), (unsigned)((Nn + 127) / 128));
            gemm_bf16_wmma<<<gg, 256, 0, stream>>>(bufA, wp, Oc, Nn, 512, Mp[l]);
        } else {
            // -------- standard variant: GEMM then spmm (M <= K) --------
            size_t atot = (size_t)Nn * (size_t)Kp[l];
            mix_convert_bf16<<<(unsigned)((atot + 255) / 256), 256, 0, stream>>>(
                hprev, mixsrc[l], bufA, Nn, Kd[l], Kp[l], hstride, (l > 0) ? 1 : 0, 0.5f);

            dim3 gg((unsigned)(Mp[l] / 64), (unsigned)((Nn + 127) / 128));
            gemm_bf16_wmma<<<gg, 256, 0, stream>>>(bufA, wp, bufM, Nn, Kp[l], Mp[l]);

            size_t on = (size_t)Nn * (size_t)Mp[l];
            zero_f32<<<8192, 256, 0, stream>>>(Oc, on);
            dim3 sg((unsigned)((E + 7) / 8), (unsigned)((Md[l] + 31) / 32));
            spmm_scatter<<<sg, dim3(32, 8), 0, stream>>>(
                evals, erow, ecol, bufM, Oc, E, Md[l], Mp[l]);
        }

        hprev = Oc;
        hstride = Mp[l];
    }

    // Row softmax over the 10 classes
    softmax10<<<(Nn + 255) / 256, 256, 0, stream>>>(hprev, (float*)d_out, Nn, Mp[4]);
}
